// EternalNeuralNetwork_50964081934643
// MI455X (gfx1250) — compile-verified, matrix operand order checked by reading, not run on hardware
//
#include <hip/hip_runtime.h>
#include <math.h>

typedef __attribute__((ext_vector_type(16))) __bf16        v16bf;
typedef __attribute__((ext_vector_type(8)))  float         v8f;
typedef __attribute__((ext_vector_type(4)))  unsigned int  v4u;
typedef __attribute__((ext_vector_type(4)))  int           v4i;

typedef __attribute__((address_space(1))) v4i* gv4i_t;   // global v4i*
typedef __attribute__((address_space(3))) v4i* lv4i_t;   // LDS v4i*

#define BM 128
#define BN 256
#define BK 64
#define LDT 72            /* BK + 8 bf16 pad -> 144B row pitch (16B aligned) */
#define A_ELEMS (BM*LDT)  /* 9216  */
#define B_ELEMS (BN*LDT)  /* 18432 */

#if __has_builtin(__builtin_amdgcn_global_load_async_to_lds_b128)
#define HAVE_ASYNC_LDS 1
#else
#define HAVE_ASYNC_LDS 0
#endif

__device__ __forceinline__ unsigned short f2bf(float f) {
  union { __bf16 b; unsigned short u; } t;
  t.b = (__bf16)f;          // RNE convert
  return t.u;
}

// 16B global -> LDS copy. Async (ASYNCcnt) on gfx1250 toolchains that expose it.
__device__ __forceinline__ void copy16_to_lds(const unsigned short* g, unsigned short* l) {
#if HAVE_ASYNC_LDS
  gv4i_t gp = (gv4i_t)(unsigned long long)g;
  lv4i_t lp = (lv4i_t)(unsigned int)(unsigned long long)l;  // low 32 bits = LDS offset
  __builtin_amdgcn_global_load_async_to_lds_b128(gp, lp, 0, 0);
#else
  *(v4u*)l = *(const v4u*)g;
#endif
}

__device__ __forceinline__ void wait_async_copies() {
#if HAVE_ASYNC_LDS
#if __has_builtin(__builtin_amdgcn_s_wait_asynccnt)
  __builtin_amdgcn_s_wait_asynccnt(0);
#else
  asm volatile("s_wait_asynccnt 0x0" ::: "memory");
#endif
#endif
}

__device__ __forceinline__ v16bf load_frag(const unsigned short* lds, int elem_off) {
  // 16-bit WMMA fragment: slots 0..7 = K[base..base+7], slots 8..15 = K[base+16..base+23]
  union { v4u u[2]; v16bf b; } t;
  t.u[0] = *(const v4u*)(lds + elem_off);
  t.u[1] = *(const v4u*)(lds + elem_off + 16);
  return t.b;
}

// ---------------------------------------------------------------------------
// GEMM: C[M][N] = act(A[M][K] (bf16) * Bt[N][K]^T (bf16) + bias (+ eternal))
// Block 128x256, 8 waves (2x4), 64x64 per wave, BK=64, double-buffered LDS.
// MODE 0: relu -> bf16; MODE 1: tanh(+eternal) -> bf16; MODE 2: relu -> f32
// ---------------------------------------------------------------------------
template<int MODE>
__global__ __launch_bounds__(256)
void gemm_bf16_kernel(const unsigned short* __restrict__ A,
                      const unsigned short* __restrict__ Bt,
                      const float* __restrict__ bias,
                      const float* __restrict__ ev,
                      unsigned short* __restrict__ outb,
                      float* __restrict__ outf,
                      int M, int N, int K)
{
  (void)M;
  __shared__ __align__(16) unsigned short Alds[2][A_ELEMS];
  __shared__ __align__(16) unsigned short Blds[2][B_ELEMS];

  const int tid  = threadIdx.x;
  const int wave = tid >> 5;
  const int lane = tid & 31;
  const int half = lane >> 4;    // 0: K[0:8]+K[16:24], 1: K[8:16]+K[24:32]
  const int lr   = lane & 15;
  const int wm   = wave >> 2;    // 2 wave-rows  (64 M each)
  const int wn   = wave & 3;     // 4 wave-cols  (64 N each)

  const int m0 = blockIdx.y * BM;
  const int n0 = blockIdx.x * BN;

  v8f acc[4][4];
  const v8f vzero = {0.f,0.f,0.f,0.f,0.f,0.f,0.f,0.f};
  #pragma unroll
  for (int mt = 0; mt < 4; ++mt)
    #pragma unroll
    for (int nt = 0; nt < 4; ++nt)
      acc[mt][nt] = vzero;

  // tile fetch: 1024 A-chunks + 2048 B-chunks of 8 bf16; 12 copies/thread
  auto issue_tile = [&](int k0, int buf) {
    #pragma unroll
    for (int i = 0; i < 4; ++i) {          // A: chunks tid + i*256 in [0,1024)
      const int c = tid + i * 256;
      const int row = c >> 3, kc = (c & 7) << 3;
      copy16_to_lds(A + (size_t)(m0 + row) * K + k0 + kc, &Alds[buf][row * LDT + kc]);
    }
    #pragma unroll
    for (int i = 0; i < 8; ++i) {          // B: chunks tid + i*256 in [0,2048)
      const int c = tid + i * 256;
      const int row = c >> 3, kc = (c & 7) << 3;
      copy16_to_lds(Bt + (size_t)(n0 + row) * K + k0 + kc, &Blds[buf][row * LDT + kc]);
    }
  };

  const int nkt = K / BK;
  issue_tile(0, 0);
  wait_async_copies();
  __syncthreads();

  for (int kt = 0; kt < nkt; ++kt) {
    const int cur = kt & 1;
    if (kt + 1 < nkt) issue_tile((kt + 1) * BK, cur ^ 1);

    const unsigned short* Ab = &Alds[cur][0];
    const unsigned short* Bb = &Blds[cur][0];
    #pragma unroll
    for (int kk = 0; kk < BK; kk += 32) {
      v16bf af[4], bfr[4];
      #pragma unroll
      for (int mt = 0; mt < 4; ++mt)
        af[mt] = load_frag(Ab, (wm * 64 + mt * 16 + lr) * LDT + kk + half * 8);
      #pragma unroll
      for (int nt = 0; nt < 4; ++nt)
        bfr[nt] = load_frag(Bb, (wn * 64 + nt * 16 + lr) * LDT + kk + half * 8);
      #pragma unroll
      for (int mt = 0; mt < 4; ++mt)
        #pragma unroll
        for (int nt = 0; nt < 4; ++nt)
          acc[mt][nt] = __builtin_amdgcn_wmma_f32_16x16x32_bf16(
              false, af[mt], false, bfr[nt], (short)0, acc[mt][nt], false, false);
    }

    wait_async_copies();   // own async writes into buf^1 complete
    __syncthreads();       // all waves done reading cur; buf^1 globally visible
  }

  // Epilogue: C layout = VGPR r -> M = half*8 + r, lane lr -> N
  const int mbase = m0 + wm * 64;
  const int nbase = n0 + wn * 64;
  #pragma unroll
  for (int mt = 0; mt < 4; ++mt) {
    #pragma unroll
    for (int nt = 0; nt < 4; ++nt) {
      const int col = nbase + nt * 16 + lr;
      float add = bias[col];
      if (MODE == 1) add += ev[col];
      const int row0 = mbase + mt * 16 + half * 8;
      #pragma unroll
      for (int r = 0; r < 8; ++r) {
        float v = acc[mt][nt][r] + add;
        v = (MODE == 1) ? tanhf(v) : fmaxf(v, 0.0f);
        const size_t idx = (size_t)(row0 + r) * N + col;
        if (MODE == 2) outf[idx] = v;
        else           outb[idx] = f2bf(v);
      }
    }
  }
}

// ---------------------------------------------------------------------------
// Eternal rotation recurrence: one workgroup per row j, 2048-float LDS vector,
// new[i] = c*x[i] - s*x[i+1] + s*x[i-1] (circular), 9*3 sweeps, out = x[0]^2.
// ---------------------------------------------------------------------------
#define ED 2048
__global__ __launch_bounds__(256)
void eternal_kernel(const float* __restrict__ ew, float* __restrict__ out)
{
  __shared__ float s[ED];
  const int j = blockIdx.x;
  const int t = threadIdx.x;
  const float init = 0.022097086912079608f; // 1/sqrt(2048)
  #pragma unroll
  for (int i = 0; i < ED / 256; ++i) s[t + i * 256] = init;
  __syncthreads();
  for (int dep = 0; dep < 9; ++dep) {
    for (int k = 0; k < 3; ++k) {
      const float ang = ew[((size_t)dep * ED + j) * ED + k];
      const float c  = cosf(ang);
      const float sn = sinf(ang);
      float nv[ED / 256];
      #pragma unroll
      for (int i = 0; i < ED / 256; ++i) {
        const int idx = t + i * 256;
        nv[i] = c * s[idx] - sn * s[(idx + 1) & (ED - 1)] + sn * s[(idx + ED - 1) & (ED - 1)];
      }
      __syncthreads();
      #pragma unroll
      for (int i = 0; i < ED / 256; ++i) s[t + i * 256] = nv[i];
      __syncthreads();
    }
  }
  if (t == 0) out[j] = s[0] * s[0];
}

// ---------------------------------------------------------------------------
// Weight transpose+convert: src[K][N] f32 -> dst[N][K] bf16 (coalesced both ways)
// ---------------------------------------------------------------------------
__global__ __launch_bounds__(256)
void wconv_kernel(const float* __restrict__ src, unsigned short* __restrict__ dst,
                  int K, int N)
{
  __shared__ float tile[32][33];
  const int bk = blockIdx.x * 32;
  const int bn = blockIdx.y * 32;
  const int tx = threadIdx.x;   // 0..31
  const int ty = threadIdx.y;   // 0..7
  #pragma unroll
  for (int r = ty; r < 32; r += 8)
    tile[r][tx] = src[(size_t)(bk + r) * N + bn + tx];
  __syncthreads();
  #pragma unroll
  for (int r = ty; r < 32; r += 8)
    dst[(size_t)(bn + r) * K + bk + tx] = f2bf(tile[tx][r]);
}

__global__ __launch_bounds__(256)
void xconv_kernel(const float* __restrict__ src, unsigned short* __restrict__ dst, int n)
{
  const int i = blockIdx.x * 256 + threadIdx.x;
  if (i < n) dst[i] = f2bf(src[i]);
}

// ---------------------------------------------------------------------------
extern "C" void kernel_launch(void* const* d_in, const int* in_sizes, int n_in,
                              void* d_out, int out_size, void* d_ws, size_t ws_size,
                              hipStream_t stream)
{
  (void)in_sizes; (void)n_in; (void)out_size; (void)ws_size;
  const float* x     = (const float*)d_in[0];
  const float* w0    = (const float*)d_in[1];
  const float* b0    = (const float*)d_in[2];
  const float* w1    = (const float*)d_in[3];
  const float* b1    = (const float*)d_in[4];
  const float* e1_ew = (const float*)d_in[5];
  const float* e1_cw = (const float*)d_in[6];
  const float* e1_cb = (const float*)d_in[7];
  const float* w2    = (const float*)d_in[8];
  const float* b2    = (const float*)d_in[9];
  const float* e2_ew = (const float*)d_in[10];
  const float* e2_cw = (const float*)d_in[11];
  const float* e2_cb = (const float*)d_in[12];
  const float* w3    = (const float*)d_in[13];
  const float* b3    = (const float*)d_in[14];

  const int Mb = 4096, IN = 1024, H = 2048, OUT = 1024;

  char* ws = (char*)d_ws;
  size_t o = 0;
  unsigned short* WT0  = (unsigned short*)(ws + o); o += (size_t)IN * H * 2;   // [H][IN]
  unsigned short* WT1  = (unsigned short*)(ws + o); o += (size_t)H * H * 2;    // [H][H]
  unsigned short* WTE1 = (unsigned short*)(ws + o); o += (size_t)H * H * 2;
  unsigned short* WT2  = (unsigned short*)(ws + o); o += (size_t)H * H * 2;
  unsigned short* WTE2 = (unsigned short*)(ws + o); o += (size_t)H * H * 2;
  unsigned short* WT3  = (unsigned short*)(ws + o); o += (size_t)H * OUT * 2;  // [OUT][H]
  unsigned short* XB   = (unsigned short*)(ws + o); o += (size_t)Mb * IN * 2;
  unsigned short* ACTA = (unsigned short*)(ws + o); o += (size_t)Mb * H * 2;
  unsigned short* ACTB = (unsigned short*)(ws + o); o += (size_t)Mb * H * 2;
  float*          EV1  = (float*)(ws + o);          o += (size_t)H * 4;
  float*          EV2  = (float*)(ws + o);          o += (size_t)H * 4;

  const dim3 tb(32, 8);
  wconv_kernel<<<dim3(IN / 32, H / 32),  tb, 0, stream>>>(w0,    WT0,  IN, H);
  wconv_kernel<<<dim3(H / 32,  H / 32),  tb, 0, stream>>>(w1,    WT1,  H,  H);
  wconv_kernel<<<dim3(H / 32,  H / 32),  tb, 0, stream>>>(e1_cw, WTE1, H,  H);
  wconv_kernel<<<dim3(H / 32,  H / 32),  tb, 0, stream>>>(w2,    WT2,  H,  H);
  wconv_kernel<<<dim3(H / 32,  H / 32),  tb, 0, stream>>>(e2_cw, WTE2, H,  H);
  wconv_kernel<<<dim3(H / 32, OUT / 32), tb, 0, stream>>>(w3,    WT3,  H,  OUT);

  xconv_kernel<<<(Mb * IN + 255) / 256, 256, 0, stream>>>(x, XB, Mb * IN);

  eternal_kernel<<<H, 256, 0, stream>>>(e1_ew, EV1);
  eternal_kernel<<<H, 256, 0, stream>>>(e2_ew, EV2);

  // L0: relu(x @ w0 + b0)  -> ACTA
  gemm_bf16_kernel<0><<<dim3(H / BN, Mb / BM), 256, 0, stream>>>(
      XB, WT0, b0, nullptr, ACTA, nullptr, Mb, H, IN);
  // L1: relu(h @ w1 + b1)  -> ACTB (h1)
  gemm_bf16_kernel<0><<<dim3(H / BN, Mb / BM), 256, 0, stream>>>(
      ACTA, WT1, b1, nullptr, ACTB, nullptr, Mb, H, H);
  // E1: tanh(h1 @ e1_cw + e1_cb + eternal1) -> ACTA (t1)
  gemm_bf16_kernel<1><<<dim3(H / BN, Mb / BM), 256, 0, stream>>>(
      ACTB, WTE1, e1_cb, EV1, ACTA, nullptr, Mb, H, H);
  // L2: relu(t1 @ w2 + b2) -> ACTB (h2)
  gemm_bf16_kernel<0><<<dim3(H / BN, Mb / BM), 256, 0, stream>>>(
      ACTA, WT2, b2, nullptr, ACTB, nullptr, Mb, H, H);
  // E2: tanh(h2 @ e2_cw + e2_cb + eternal2) -> ACTA (t2)
  gemm_bf16_kernel<1><<<dim3(H / BN, Mb / BM), 256, 0, stream>>>(
      ACTB, WTE2, e2_cb, EV2, ACTA, nullptr, Mb, H, H);
  // L3: relu(t2 @ w3 + b3) -> d_out (f32)
  gemm_bf16_kernel<2><<<dim3(OUT / BN, Mb / BM), 256, 0, stream>>>(
      ACTA, WT3, b3, nullptr, nullptr, (float*)d_out, Mb, OUT, H);
}
